// DecoderRNN_36962488549527
// MI455X (gfx1250) — matrix operand order repaired and checked by reading, not compiled, not run on hardware
//
#include <hip/hip_runtime.h>
#include <hip/hip_bf16.h>

#define B      64
#define ENC_T  1500
#define ENC_D  512
#define H      512
#define ATTN   512
#define VOCAB  2000
#define MAX_LEN 100
#define SOS    1

typedef __bf16 bf16_t;
typedef __attribute__((ext_vector_type(16))) __bf16 v16bf;
typedef __attribute__((ext_vector_type(8)))  __bf16 v8bf;
typedef __attribute__((ext_vector_type(8)))  float  v8f;

// ---------------------------------------------------------------------------
// WMMA operand loaders for v_wmma_f32_16x16x32_bf16 (wave32).
// A (16x32, MxK): lane<16 -> M=lane, K = k0..k0+15
//                 lane>=16 -> same M, K pattern shifted by +16.
// B (32x16, KxN): lane holds N=lane&15, 16 contiguous K values,
//                 upper half-lanes shifted by +16.
// Both GEMMs here are X @ W^T with row-major W (N x K), so A and B load
// straight from row-major storage with two 16-byte loads each.
// ---------------------------------------------------------------------------
__device__ __forceinline__ v16bf load_a_bf16(const bf16_t* base, int ld,
                                             int m0, int k0, int lane) {
  int r  = m0 + (lane & 15);
  int kb = k0 + (lane & 16);                   // +16 for upper half-lanes
  const bf16_t* p = base + (size_t)r * ld + kb;
  v8bf lo = *(const v8bf*)p;                   // K: kb .. kb+7
  v8bf hi = *(const v8bf*)(p + 8);             // K: kb+8 .. kb+15
  return __builtin_shufflevector(lo, hi, 0,1,2,3,4,5,6,7,8,9,10,11,12,13,14,15);
}

__device__ __forceinline__ v16bf load_b_bf16(const bf16_t* base, int ld,
                                             int n0, int k0, int lane) {
  int c  = n0 + (lane & 15);
  int kb = k0 + (lane & 16);                   // +16 for upper half-lanes
  const bf16_t* p = base + (size_t)c * ld + kb;
  v8bf lo = *(const v8bf*)p;                   // K: kb .. kb+7
  v8bf hi = *(const v8bf*)(p + 8);             // K: kb+8 .. kb+15
  return __builtin_shufflevector(lo, hi, 0,1,2,3,4,5,6,7,8,9,10,11,12,13,14,15);
}

// C = X(MxK) @ W(NxK)^T + bias, f32 accumulate/output. One wave per 16x16 tile.
__global__ void gemm_xwT_f32(const bf16_t* __restrict__ X, int ldx,
                             const bf16_t* __restrict__ W, int ldw,
                             const float* __restrict__ bias,
                             float* __restrict__ C, int ldc,
                             int M, int N, int K) {
  int wave = (int)((blockIdx.x * blockDim.x + threadIdx.x) >> 5);
  int lane = threadIdx.x & 31;
  int ntiles = N >> 4;
  int total  = (M >> 4) * ntiles;
  if (wave >= total) return;
  int m0 = (wave / ntiles) << 4;
  int n0 = (wave % ntiles) << 4;
  v8f acc = {0.f,0.f,0.f,0.f,0.f,0.f,0.f,0.f};
  for (int k0 = 0; k0 < K; k0 += 32) {
    v16bf a = load_a_bf16(X, ldx, m0, k0, lane);
    v16bf b = load_b_bf16(W, ldw, n0, k0, lane);
    acc = __builtin_amdgcn_wmma_f32_16x16x32_bf16(false, a, false, b,
                                                  (short)0, acc, false, false);
  }
  int n  = n0 + (lane & 15);
  float bv = bias ? bias[n] : 0.0f;
  int mb = m0 + ((lane & 16) >> 1);            // VGPR r holds M = mb + r
#pragma unroll
  for (int r = 0; r < 8; ++r)
    C[(size_t)(mb + r) * ldc + n] = acc[r] + bv;
}

// Same GEMM but bf16 output (used for the big one-time enc_proj precompute,
// keeping the 98 MB streaming tensor L2-resident in bf16).
__global__ void gemm_xwT_bf16(const bf16_t* __restrict__ X, int ldx,
                              const bf16_t* __restrict__ W, int ldw,
                              bf16_t* __restrict__ C, int ldc,
                              int M, int N, int K) {
  int wave = (int)((blockIdx.x * blockDim.x + threadIdx.x) >> 5);
  int lane = threadIdx.x & 31;
  int ntiles = N >> 4;
  long total = (long)(M >> 4) * ntiles;
  if (wave >= total) return;
  int m0 = (int)(wave / ntiles) << 4;
  int n0 = (int)(wave % ntiles) << 4;
  v8f acc = {0.f,0.f,0.f,0.f,0.f,0.f,0.f,0.f};
  for (int k0 = 0; k0 < K; k0 += 32) {
    v16bf a = load_a_bf16(X, ldx, m0, k0, lane);
    v16bf b = load_b_bf16(W, ldw, n0, k0, lane);
    acc = __builtin_amdgcn_wmma_f32_16x16x32_bf16(false, a, false, b,
                                                  (short)0, acc, false, false);
  }
  int n  = n0 + (lane & 15);
  int mb = m0 + ((lane & 16) >> 1);
#pragma unroll
  for (int r = 0; r < 8; ++r)
    C[(size_t)(mb + r) * ldc + n] = (bf16_t)acc[r];
}

// ---------------------------------------------------------------------------
__global__ void cvt_bf16_kernel(const float* __restrict__ src,
                                bf16_t* __restrict__ dst, long n) {
  long i = (long)blockIdx.x * blockDim.x + threadIdx.x;
  if (i < n) dst[i] = (bf16_t)src[i];
}

__global__ void init_state_kernel(float* h, bf16_t* hb, float* ctx,
                                  float* aw, int* tok) {
  int i = blockIdx.x * blockDim.x + threadIdx.x;
  if (i < B * H)     { h[i] = 0.f; hb[i] = (bf16_t)0.f; ctx[i] = 0.f; }
  if (i < B * ENC_T) { aw[i] = 0.f; }
  if (i < B)         { tok[i] = SOS; }
}

// xc = [emb[tok], ctx] in bf16  (GRU input row, 64 x 1024)
__global__ void prep_xc_kernel(const bf16_t* __restrict__ emb_bf,
                               const int* __restrict__ tok,
                               const float* __restrict__ ctx,
                               bf16_t* __restrict__ xc) {
  int b = blockIdx.x, i = threadIdx.x;   // blockDim = 512
  int t = tok[b];
  xc[(size_t)b * (H + ENC_D) + i]     = emb_bf[(size_t)t * H + i];
  xc[(size_t)b * (H + ENC_D) + H + i] = (bf16_t)ctx[(size_t)b * ENC_D + i];
}

// GRU gate nonlinearity: h_new = (1-z)*n + z*h  (PyTorch GRU math)
__global__ void gru_gate_kernel(const float* __restrict__ gi,
                                const float* __restrict__ gh,
                                float* __restrict__ h,
                                bf16_t* __restrict__ hb,
                                bf16_t* __restrict__ hc) {
  int b = blockIdx.x, j = threadIdx.x;   // blockDim = 512
  size_t o = (size_t)b * 3 * H;
  float ir = gi[o + j], iz = gi[o + H + j], in_ = gi[o + 2 * H + j];
  float hr = gh[o + j], hz = gh[o + H + j], hn  = gh[o + 2 * H + j];
  float r = 1.f / (1.f + expf(-(ir + hr)));
  float z = 1.f / (1.f + expf(-(iz + hz)));
  float n = tanhf(in_ + r * hn);
  float hnew = (1.f - z) * n + z * h[(size_t)b * H + j];
  h[(size_t)b * H + j]  = hnew;
  hb[(size_t)b * H + j] = (bf16_t)hnew;
  hc[(size_t)b * (H + ENC_D) + j] = (bf16_t)hnew;   // logits concat, col 0..511
}

// score[b,t] = fc_w . tanh(q[b,:] + enc_proj[b,t,:] + conv[b,t,:] + bias) + fc_b
// conv[b,t,a] = conv_b[a] + w0[a]*aw[t-1] + w1[a]*aw[t] + w2[a]*aw[t+1]
__global__ void energy_kernel(const float* __restrict__ q,
                              const bf16_t* __restrict__ enc_proj,
                              const float* __restrict__ aw,
                              const float* __restrict__ conv_w,
                              const float* __restrict__ conv_b,
                              const float* __restrict__ attn_bias,
                              const float* __restrict__ fc_w,
                              const float* __restrict__ fc_b,
                              float* __restrict__ score) {
  __shared__ float sq[ATTN], sw0[ATTN], sw1[ATTN], sw2[ATTN], sfw[ATTN];
  int b = blockIdx.x;
  int tid = threadIdx.x;
  for (int a = tid; a < ATTN; a += blockDim.x) {
    sq[a]  = q[(size_t)b * ATTN + a] + attn_bias[a] + conv_b[a];
    sw0[a] = conv_w[a * 3 + 0];
    sw1[a] = conv_w[a * 3 + 1];
    sw2[a] = conv_w[a * 3 + 2];
    sfw[a] = fc_w[a];
  }
  __syncthreads();
  int t = blockIdx.y * blockDim.x + tid;
  if (t >= ENC_T) return;
  float am1 = (t > 0)         ? aw[(size_t)b * ENC_T + t - 1] : 0.f;
  float a0  =                   aw[(size_t)b * ENC_T + t];
  float ap1 = (t < ENC_T - 1) ? aw[(size_t)b * ENC_T + t + 1] : 0.f;
  const v8bf* pp = (const v8bf*)(enc_proj + ((size_t)b * ENC_T + t) * ATTN);
  float s = 0.f;
  for (int c = 0; c < ATTN / 8; ++c) {
    v8bf v = pp[c];
#pragma unroll
    for (int j = 0; j < 8; ++j) {
      int a = c * 8 + j;
      float e = sq[a] + sw0[a] * am1 + sw1[a] * a0 + sw2[a] * ap1 + (float)v[j];
      s += tanhf(e) * sfw[a];
    }
  }
  score[(size_t)b * ENC_T + t] = s + fc_b[0];
}

// softmax over T, then ctx[b,:] = aw @ encoder_outputs[b]  (bf16 stream)
__global__ void softmax_ctx_kernel(const float* __restrict__ score,
                                   const bf16_t* __restrict__ enc_bf,
                                   float* __restrict__ aw,
                                   float* __restrict__ ctx,
                                   bf16_t* __restrict__ hc) {
  __shared__ float red[256];
  __shared__ float awsh[ENC_T];
  int b = blockIdx.x, tid = threadIdx.x;
  const float* sr = score + (size_t)b * ENC_T;
  float m = -1e30f;
  for (int t = tid; t < ENC_T; t += 256) m = fmaxf(m, sr[t]);
  red[tid] = m; __syncthreads();
  for (int s = 128; s > 0; s >>= 1) {
    if (tid < s) red[tid] = fmaxf(red[tid], red[tid + s]);
    __syncthreads();
  }
  m = red[0]; __syncthreads();
  float sum = 0.f;
  for (int t = tid; t < ENC_T; t += 256) {
    float e = expf(sr[t] - m);
    awsh[t] = e; sum += e;
  }
  red[tid] = sum; __syncthreads();
  for (int s = 128; s > 0; s >>= 1) {
    if (tid < s) red[tid] += red[tid + s];
    __syncthreads();
  }
  float inv = 1.f / red[0];
  __syncthreads();
  for (int t = tid; t < ENC_T; t += 256) {
    float w = awsh[t] * inv;
    awsh[t] = w;
    aw[(size_t)b * ENC_T + t] = w;
  }
  __syncthreads();
  for (int e = tid; e < ENC_D; e += 256) {
    float acc = 0.f;
    const bf16_t* pe = enc_bf + (size_t)b * ENC_T * ENC_D + e;
    for (int t = 0; t < ENC_T; ++t) acc += awsh[t] * (float)pe[(size_t)t * ENC_D];
    ctx[(size_t)b * ENC_D + e] = acc;
    hc[(size_t)b * (H + ENC_D) + H + e] = (bf16_t)acc;  // logits concat, col 512..
  }
}

// log_softmax over V -> d_out[b, step, :]; greedy argmax -> tok (first-max ties)
__global__ void lsm_argmax_kernel(const float* __restrict__ logits,
                                  float* __restrict__ out, int step,
                                  int* __restrict__ tok) {
  __shared__ float rv[256];
  __shared__ int   ri[256];
  int b = blockIdx.x, tid = threadIdx.x;
  const float* lr = logits + (size_t)b * VOCAB;
  float m = -1e30f; int mi = 0;
  for (int v = tid; v < VOCAB; v += 256) {
    float x = lr[v];
    if (x > m) { m = x; mi = v; }
  }
  rv[tid] = m; ri[tid] = mi; __syncthreads();
  for (int s = 128; s > 0; s >>= 1) {
    if (tid < s) {
      if (rv[tid + s] > rv[tid] ||
          (rv[tid + s] == rv[tid] && ri[tid + s] < ri[tid])) {
        rv[tid] = rv[tid + s]; ri[tid] = ri[tid + s];
      }
    }
    __syncthreads();
  }
  m = rv[0];
  if (tid == 0) tok[b] = ri[0];
  __syncthreads();
  float sum = 0.f;
  for (int v = tid; v < VOCAB; v += 256) sum += expf(lr[v] - m);
  rv[tid] = sum; __syncthreads();
  for (int s = 128; s > 0; s >>= 1) {
    if (tid < s) rv[tid] += rv[tid + s];
    __syncthreads();
  }
  float lse = m + logf(rv[0]);
  float* orow = out + ((size_t)b * MAX_LEN + step) * VOCAB;
  for (int v = tid; v < VOCAB; v += 256) orow[v] = lr[v] - lse;
}

// ---------------------------------------------------------------------------
extern "C" void kernel_launch(void* const* d_in, const int* in_sizes, int n_in,
                              void* d_out, int out_size, void* d_ws, size_t ws_size,
                              hipStream_t stream) {
  (void)in_sizes; (void)n_in; (void)out_size; (void)ws_size;
  const float* enc    = (const float*)d_in[0];
  const float* emb    = (const float*)d_in[1];
  const float* w_ih   = (const float*)d_in[2];
  const float* w_hh   = (const float*)d_in[3];
  const float* b_ih   = (const float*)d_in[4];
  const float* b_hh   = (const float*)d_in[5];
  const float* conv_w = (const float*)d_in[6];
  const float* conv_b = (const float*)d_in[7];
  const float* wq     = (const float*)d_in[8];
  const float* attn_v = (const float*)d_in[9];
  const float* fc_w   = (const float*)d_in[10];
  const float* fc_b   = (const float*)d_in[11];
  const float* abias  = (const float*)d_in[12];
  const float* out_w  = (const float*)d_in[13];
  const float* out_b  = (const float*)d_in[14];
  float* out = (float*)d_out;

  char* p = (char*)d_ws;
  auto alloc = [&](size_t bytes) -> char* {
    char* r = p; p += (bytes + 255) & ~(size_t)255; return r;
  };
  bf16_t* enc_bf   = (bf16_t*)alloc((size_t)B * ENC_T * ENC_D * 2);
  bf16_t* eproj_bf = (bf16_t*)alloc((size_t)B * ENC_T * ATTN * 2);
  bf16_t* wih_bf   = (bf16_t*)alloc((size_t)3 * H * (H + ENC_D) * 2);
  bf16_t* whh_bf   = (bf16_t*)alloc((size_t)3 * H * H * 2);
  bf16_t* wq_bf    = (bf16_t*)alloc((size_t)ATTN * H * 2);
  bf16_t* av_bf    = (bf16_t*)alloc((size_t)ATTN * ENC_D * 2);
  bf16_t* ow_bf    = (bf16_t*)alloc((size_t)VOCAB * (H + ENC_D) * 2);
  bf16_t* emb_bf   = (bf16_t*)alloc((size_t)VOCAB * H * 2);
  bf16_t* xc_bf    = (bf16_t*)alloc((size_t)B * (H + ENC_D) * 2);
  bf16_t* hc_bf    = (bf16_t*)alloc((size_t)B * (H + ENC_D) * 2);
  bf16_t* h_bf     = (bf16_t*)alloc((size_t)B * H * 2);
  float* h_f       = (float*)alloc((size_t)B * H * 4);
  float* ctx_f     = (float*)alloc((size_t)B * ENC_D * 4);
  float* gi        = (float*)alloc((size_t)B * 3 * H * 4);
  float* gh        = (float*)alloc((size_t)B * 3 * H * 4);
  float* q_f       = (float*)alloc((size_t)B * ATTN * 4);
  float* aw        = (float*)alloc((size_t)B * ENC_T * 4);
  float* scoref    = (float*)alloc((size_t)B * ENC_T * 4);
  float* logits    = (float*)alloc((size_t)B * VOCAB * 4);
  int*   tok       = (int*)alloc((size_t)B * 4);

  auto cvt = [&](const float* s, bf16_t* d, long n) {
    cvt_bf16_kernel<<<(unsigned)((n + 255) / 256), 256, 0, stream>>>(s, d, n);
  };
  cvt(enc,    enc_bf, (long)B * ENC_T * ENC_D);
  cvt(w_ih,   wih_bf, (long)3 * H * (H + ENC_D));
  cvt(w_hh,   whh_bf, (long)3 * H * H);
  cvt(wq,     wq_bf,  (long)ATTN * H);
  cvt(attn_v, av_bf,  (long)ATTN * ENC_D);
  cvt(out_w,  ow_bf,  (long)VOCAB * (H + ENC_D));
  cvt(emb,    emb_bf, (long)VOCAB * H);

  init_state_kernel<<<(B * ENC_T + 255) / 256, 256, 0, stream>>>(h_f, h_bf,
                                                                 ctx_f, aw, tok);

  // One-time enc_proj = encoder_outputs @ attn_v^T  (96000 x 512, bf16 out)
  {
    long tiles = ((long)B * ENC_T / 16) * (ATTN / 16);
    gemm_xwT_bf16<<<(unsigned)((tiles * 32 + 255) / 256), 256, 0, stream>>>(
        enc_bf, ENC_D, av_bf, ENC_D, eproj_bf, ATTN, B * ENC_T, ATTN, ENC_D);
  }

  auto gemm = [&](const bf16_t* X, int ldx, const bf16_t* W, int ldw,
                  const float* bias, float* C, int ldc, int M, int N, int K) {
    long tiles = (long)(M / 16) * (N / 16);
    gemm_xwT_f32<<<(unsigned)((tiles * 32 + 255) / 256), 256, 0, stream>>>(
        X, ldx, W, ldw, bias, C, ldc, M, N, K);
  };

  for (int step = 0; step < MAX_LEN; ++step) {
    prep_xc_kernel<<<B, 512, 0, stream>>>(emb_bf, tok, ctx_f, xc_bf);
    gemm(xc_bf, H + ENC_D, wih_bf, H + ENC_D, b_ih, gi, 3 * H, B, 3 * H, H + ENC_D);
    gemm(h_bf, H, whh_bf, H, b_hh, gh, 3 * H, B, 3 * H, H);
    gru_gate_kernel<<<B, H, 0, stream>>>(gi, gh, h_f, h_bf, hc_bf);
    gemm(h_bf, H, wq_bf, H, nullptr, q_f, ATTN, B, ATTN, H);
    dim3 eg(B, (ENC_T + 255) / 256);
    energy_kernel<<<eg, 256, 0, stream>>>(q_f, eproj_bf, aw, conv_w, conv_b,
                                          abias, fc_w, fc_b, scoref);
    softmax_ctx_kernel<<<B, 256, 0, stream>>>(scoref, enc_bf, aw, ctx_f, hc_bf);
    gemm(hc_bf, H + ENC_D, ow_bf, H + ENC_D, out_b, logits, VOCAB, B, VOCAB,
         H + ENC_D);
    lsm_argmax_kernel<<<B, 256, 0, stream>>>(logits, out, step, tok);
  }
}
